// MSTHN_21199958573608
// MI455X (gfx1250) — matrix operand order repaired and checked by reading, not compiled, not run on hardware
//
#include <hip/hip_runtime.h>
#include <hip/hip_bf16.h>
#include <math.h>

typedef __attribute__((ext_vector_type(16))) _Float16 v16h;
typedef __attribute__((ext_vector_type(8)))  float    v8f;

#define SEQ     100
#define EMBD    128
#define NHEADS  8
#define HD      16
#define NB      1024
#define NUSERS  10000
#define NPOIS   50000
#define NNODES  60001
#define NNZG    1920000L
#define NNZH    1600000L

// ---------------------------------------------------------------------------
// WMMA GEMM: C[M,N] = act(A[M,K] @ W[N,K]^T + bias)
// Guard-free inner loop: M (compute), N, K padded to tile multiples by caller
// (128 rows per block, 32 cols per wave, K%32==0). GUARD=1 adds a store-row
// guard against Mstore (only needed for the padded geo bmm).
// Each wave computes a 16x32 tile: one A fragment feeds two WMMAs.
// Batched via blockIdx.z (element strides sA/sW/sC).
// ACT: 0 none, 1 relu, 2 tanh, 3 sigmoid(v + extra[(row/SEQ)*N + col])
// ---------------------------------------------------------------------------
template<int ACT, int BIAS, int GUARD>
__global__ void wmma_gemm_kernel(const float* __restrict__ A, const float* __restrict__ W,
                                 const float* __restrict__ bias, float* __restrict__ C,
                                 int Mstore, int N, int K, int lda, int ldw, int ldc,
                                 long sA, long sW, long sC, const float* __restrict__ extra)
{
    const int lane  = threadIdx.x & 31;
    const int wave  = threadIdx.x >> 5;
    const int tileN = blockIdx.x * 32;
    const int tileM = (blockIdx.y * 8 + wave) * 16;

    const int hiA = (lane & 16) ? 8 : 0;    // C frag: upper half-wave holds rows M=8..15
    const int hiB = (lane & 16) ? 16 : 0;   // B frag: upper half-wave covers K=16..31

    const float* Ab = A + (long)blockIdx.z * sA + (long)(tileM + (lane & 15)) * lda;
    const float* W0 = W + (long)blockIdx.z * sW + (long)(tileN + (lane & 15)) * ldw;
    const float* W1 = W0 + 16L * (long)ldw;

    v8f c0 = {}, c1 = {};
    for (int k0 = 0; k0 < K; k0 += 32) {
        // A 16x32 f16 fragment: elements 0..15 -> K=hiB..hiB+15
        const float4* ap = (const float4*)(Ab + k0 + hiB);
        float4 a0 = ap[0], a1 = ap[1], a2 = ap[2], a3 = ap[3];
        v16h af;
        af[0]=(_Float16)a0.x; af[1]=(_Float16)a0.y; af[2]=(_Float16)a0.z; af[3]=(_Float16)a0.w;
        af[4]=(_Float16)a1.x; af[5]=(_Float16)a1.y; af[6]=(_Float16)a1.z; af[7]=(_Float16)a1.w;
        af[8]=(_Float16)a2.x; af[9]=(_Float16)a2.y; af[10]=(_Float16)a2.z; af[11]=(_Float16)a2.w;
        af[12]=(_Float16)a3.x; af[13]=(_Float16)a3.y; af[14]=(_Float16)a3.z; af[15]=(_Float16)a3.w;

        // B 32x16 f16 fragment (from W row-major): elements 0..15 -> K=hiB..hiB+15
        const float4* bp0 = (const float4*)(W0 + k0 + hiB);
        const float4* bp1 = (const float4*)(W1 + k0 + hiB);
        float4 b0 = bp0[0], b1 = bp0[1], b2 = bp0[2], b3 = bp0[3];
        float4 d0 = bp1[0], d1 = bp1[1], d2 = bp1[2], d3 = bp1[3];
        v16h bf0, bf1;
        bf0[0]=(_Float16)b0.x; bf0[1]=(_Float16)b0.y; bf0[2]=(_Float16)b0.z; bf0[3]=(_Float16)b0.w;
        bf0[4]=(_Float16)b1.x; bf0[5]=(_Float16)b1.y; bf0[6]=(_Float16)b1.z; bf0[7]=(_Float16)b1.w;
        bf0[8]=(_Float16)b2.x; bf0[9]=(_Float16)b2.y; bf0[10]=(_Float16)b2.z; bf0[11]=(_Float16)b2.w;
        bf0[12]=(_Float16)b3.x; bf0[13]=(_Float16)b3.y; bf0[14]=(_Float16)b3.z; bf0[15]=(_Float16)b3.w;
        bf1[0]=(_Float16)d0.x; bf1[1]=(_Float16)d0.y; bf1[2]=(_Float16)d0.z; bf1[3]=(_Float16)d0.w;
        bf1[4]=(_Float16)d1.x; bf1[5]=(_Float16)d1.y; bf1[6]=(_Float16)d1.z; bf1[7]=(_Float16)d1.w;
        bf1[8]=(_Float16)d2.x; bf1[9]=(_Float16)d2.y; bf1[10]=(_Float16)d2.z; bf1[11]=(_Float16)d2.w;
        bf1[12]=(_Float16)d3.x; bf1[13]=(_Float16)d3.y; bf1[14]=(_Float16)d3.z; bf1[15]=(_Float16)d3.w;

        c0 = __builtin_amdgcn_wmma_f32_16x16x32_f16(false, af, false, bf0, (short)0, c0, false, false);
        c1 = __builtin_amdgcn_wmma_f32_16x16x32_f16(false, af, false, bf1, (short)0, c1, false, false);
    }

    // ---- epilogue: row-invariant terms hoisted, straight-line when GUARD==0
    const int col0 = tileN + (lane & 15);
    const int col1 = col0 + 16;
    const float bias0 = BIAS ? bias[col0] : 0.0f;
    const float bias1 = BIAS ? bias[col1] : 0.0f;
    float* Cp = C + (long)blockIdx.z * sC + (long)(tileM + hiA) * ldc + col0;

    #pragma unroll
    for (int r = 0; r < 8; ++r) {
        const int row = tileM + r + hiA;    // C layout: VGPR r -> M=r (lo) / M=8+r (hi)
        if (!GUARD || row < Mstore) {
            float v0 = c0[r] + bias0;
            float v1 = c1[r] + bias1;
            if (ACT == 1) {
                v0 = fmaxf(v0, 0.0f); v1 = fmaxf(v1, 0.0f);
            } else if (ACT == 2) {
                v0 = tanhf(v0); v1 = tanhf(v1);
            } else if (ACT == 3) {
                const float* ex = extra + (long)(row / SEQ) * N;
                v0 = 1.0f / (1.0f + expf(-(v0 + ex[col0])));
                v1 = 1.0f / (1.0f + expf(-(v1 + ex[col1])));
            }
            Cp[(long)r * ldc]      = v0;
            Cp[(long)r * ldc + 16] = v1;
        }
    }
}

// ---------------------------------------------------------------------------
// Elementwise / data movement kernels
// ---------------------------------------------------------------------------
__global__ void init2_kernel(float* a, float* b, const float* src, long n) {
    long t = (long)blockIdx.x * blockDim.x + threadIdx.x;
    if (t >= n) return;
    float v = src[t]; a[t] = v; b[t] = v;
}
__global__ void zero_kernel(float* p, long n) {
    long t = (long)blockIdx.x * blockDim.x + threadIdx.x;
    if (t < n) p[t] = 0.0f;
}
__global__ void copy_kernel(float* dst, const float* src, long n) {
    long t = (long)blockIdx.x * blockDim.x + threadIdx.x;
    if (t < n) dst[t] = src[t];
}
__global__ void add_kernel(float* dst, const float* src, long n) {
    long t = (long)blockIdx.x * blockDim.x + threadIdx.x;
    if (t < n) dst[t] += src[t];
}

// adj_pad[b, r(128), k(128)] = adj[b, r, k] zero-padded from 100x100
__global__ void adjpad_kernel(const float* __restrict__ adj, float* __restrict__ pad) {
    long t = (long)blockIdx.x * blockDim.x + threadIdx.x;
    long total = (long)NB * 128 * 128;
    if (t >= total) return;
    long b = t >> 14; int r = (int)((t >> 7) & 127); int k = (int)(t & 127);
    pad[t] = (r < SEQ && k < SEQ) ? adj[((long)b * SEQ + r) * SEQ + k] : 0.0f;
}

// seq_padT[b, d(128), s(128)] = seq_e[b, s, d] zero-padded over s
__global__ void seqT_kernel(const float* __restrict__ seq_e, float* __restrict__ padT) {
    long t = (long)blockIdx.x * blockDim.x + threadIdx.x;
    long total = (long)NB * 128 * 128;
    if (t >= total) return;
    long b = t >> 14; int d = (int)((t >> 7) & 127); int s = (int)(t & 127);
    padT[t] = (s < SEQ) ? seq_e[((long)b * SEQ + s) * EMBD + d] : 0.0f;
}

// dst[r,:] = src[idx[r],:]
__global__ void gather_rows_kernel(const float* __restrict__ src, const int* __restrict__ idx,
                                   float* __restrict__ dst, long n_rows) {
    long t = (long)blockIdx.x * blockDim.x + threadIdx.x;
    long total = n_rows * EMBD;
    if (t >= total) return;
    long r = t >> 7; int d = (int)(t & 127);
    dst[t] = src[(long)idx[r] * EMBD + d];
}

// tot = seq_e + geo_e + pos_local[l+1]
__global__ void tot_kernel(const float* __restrict__ seq_e, const float* __restrict__ geo_e,
                           const float* __restrict__ pos_local, float* __restrict__ tot) {
    long t = (long)blockIdx.x * blockDim.x + threadIdx.x;
    long total = (long)NB * SEQ * EMBD;
    if (t >= total) return;
    long bl = t >> 7; int d = (int)(t & 127); int l = (int)(bl % SEQ);
    tot[t] = seq_e[t] + geo_e[t] + pos_local[(long)(l + 1) * EMBD + d];
}

// per-(batch,head) softmax attention; K/V staged in LDS
__global__ void attention_kernel(const float* __restrict__ qkv, float* __restrict__ out) {
    __shared__ float ks[SEQ][HD];
    __shared__ float vs[SEQ][HD];
    int bh = blockIdx.x;
    int b = bh / NHEADS, h = bh % NHEADS;
    const float* base = qkv + (long)b * SEQ * 3 * EMBD + h * HD;
    for (int t = threadIdx.x; t < SEQ * HD; t += blockDim.x) {
        int l = t / HD, d = t % HD;
        ks[l][d] = base[(long)l * 3 * EMBD + EMBD + d];
        vs[l][d] = base[(long)l * 3 * EMBD + 2 * EMBD + d];
    }
    __syncthreads();
    int i = threadIdx.x;
    if (i < SEQ) {
        float q[HD];
        #pragma unroll
        for (int d = 0; d < HD; ++d) q[d] = base[(long)i * 3 * EMBD + d];
        float m = -1e30f;
        for (int j = 0; j < SEQ; ++j) {
            float s = 0.0f;
            #pragma unroll
            for (int d = 0; d < HD; ++d) s += q[d] * ks[j][d];
            m = fmaxf(m, s * 0.25f);
        }
        float sum = 0.0f, o[HD];
        #pragma unroll
        for (int d = 0; d < HD; ++d) o[d] = 0.0f;
        for (int j = 0; j < SEQ; ++j) {
            float s = 0.0f;
            #pragma unroll
            for (int d = 0; d < HD; ++d) s += q[d] * ks[j][d];
            float e = expf(s * 0.25f - m);
            sum += e;
            #pragma unroll
            for (int d = 0; d < HD; ++d) o[d] += e * vs[j][d];
        }
        float inv = 1.0f / sum;
        float* op = out + ((long)b * SEQ + i) * EMBD + h * HD;
        #pragma unroll
        for (int d = 0; d < HD; ++d) op[d] = o[d] * inv;
    }
}

// users[b] = mean_l mha[b,l,:] scattered into nodes[idx[b]]
__global__ void users_kernel(const float* __restrict__ mha, const int* __restrict__ uidx,
                             float* __restrict__ nodes) {
    int b = blockIdx.x, d = threadIdx.x;
    float s = 0.0f;
    for (int l = 0; l < SEQ; ++l) s += mha[((long)b * SEQ + l) * EMBD + d];
    nodes[(long)uidx[b] * EMBD + d] = s * (1.0f / SEQ);
}

// y[rows[e],:] += vals[e] * x[cols[e],:]  (30 MB accumulator lives in L2)
__global__ void spmm_kernel(const int* __restrict__ rows, const int* __restrict__ cols,
                            const float* __restrict__ vals, const float* __restrict__ x,
                            float* __restrict__ y, long nnz) {
    long t = (long)blockIdx.x * blockDim.x + threadIdx.x;
    long total = nnz * EMBD;
    if (t >= total) return;
    long e = t >> 7; int d = (int)(t & 127);
    atomicAdd(&y[(long)rows[e] * EMBD + d], vals[e] * x[(long)cols[e] * EMBD + d]);
}

// nodes[:-1] = core; acc += nodes
__global__ void commit_kernel(float* __restrict__ nodes, float* __restrict__ acc,
                              const float* __restrict__ core, long n) {
    long t = (long)blockIdx.x * blockDim.x + threadIdx.x;
    if (t >= n) return;
    float v = (t < (long)(NNODES - 1) * EMBD) ? core[t] : nodes[t];
    nodes[t] = v;
    acc[t] += v;
}

// fusion = [acc[:NU]/3, acc[NU:NU+NP]/3 + accg/3, 0]; also emits pois output
__global__ void fusion_kernel(const float* __restrict__ acc, const float* __restrict__ accg,
                              float* __restrict__ fusion, float* __restrict__ out_pois, long n) {
    long t = (long)blockIdx.x * blockDim.x + threadIdx.x;
    if (t >= n) return;
    long r = t >> 7;
    float f;
    if (r < NUSERS) f = acc[t] * (1.0f / 3.0f);
    else if (r < NUSERS + NPOIS) {
        long j = t - (long)NUSERS * EMBD;
        f = acc[t] * (1.0f / 3.0f) + accg[j] * (1.0f / 3.0f);
        out_pois[j] = f;
    } else f = 0.0f;
    fusion[t] = f;
}

__global__ void hs_kernel(const float* __restrict__ seq_h, const float* __restrict__ lens,
                          float* __restrict__ hs) {
    int b = blockIdx.x, d = threadIdx.x;
    float s = 0.0f;
    for (int l = 0; l < SEQ; ++l) s += seq_h[((long)b * SEQ + l) * EMBD + d];
    hs[(long)b * EMBD + d] = s / lens[b];
}

// cat[b,l] = [pos_emb_glob[(l+1)*mask[b,l]], seq_h[b,l]]
__global__ void cat_kernel(const float* __restrict__ pos_glob, const int* __restrict__ mask,
                           const float* __restrict__ seq_h, float* __restrict__ cat) {
    long t = (long)blockIdx.x * blockDim.x + threadIdx.x;
    long total = (long)NB * SEQ * EMBD;
    if (t >= total) return;
    long bl = t >> 7; int d = (int)(t & 127); int l = (int)(bl % SEQ);
    int pidx = (l + 1) * mask[bl];
    cat[bl * 256 + d]       = pos_glob[(long)pidx * EMBD + d];
    cat[bl * 256 + 128 + d] = seq_h[t];
}

__global__ void beta_kernel(const float* __restrict__ nh, const float* __restrict__ w2,
                            float* __restrict__ beta, long bl_total) {
    long t = (long)blockIdx.x * blockDim.x + threadIdx.x;
    if (t >= bl_total) return;
    float s = 0.0f;
    for (int k = 0; k < EMBD; ++k) s += nh[t * EMBD + k] * w2[k];
    beta[t] = s;
}

// out_users[b] = sum_l beta[b,l]*seq_h[b,l,:] + acc[uidx[b]]/3
__global__ void select_kernel(const float* __restrict__ beta, const float* __restrict__ seq_h,
                              const float* __restrict__ acc, const int* __restrict__ uidx,
                              float* __restrict__ out) {
    int b = blockIdx.x, d = threadIdx.x;
    float s = 0.0f;
    for (int l = 0; l < SEQ; ++l)
        s += beta[(long)b * SEQ + l] * seq_h[((long)b * SEQ + l) * EMBD + d];
    out[(long)b * EMBD + d] = s + acc[(long)uidx[b] * EMBD + d] * (1.0f / 3.0f);
}

// ---------------------------------------------------------------------------
extern "C" void kernel_launch(void* const* d_in, const int* in_sizes, int n_in,
                              void* d_out, int out_size, void* d_ws, size_t ws_size,
                              hipStream_t stream) {
    const float* nodes_emb   = (const float*)d_in[0];
    const float* pos_local   = (const float*)d_in[1];
    const float* fc_geo_w    = (const float*)d_in[2];
    const float* fc_geo_b    = (const float*)d_in[3];
    const float* in_proj_w   = (const float*)d_in[4];
    const float* in_proj_b   = (const float*)d_in[5];
    const float* out_proj_w  = (const float*)d_in[6];
    const float* out_proj_b  = (const float*)d_in[7];
    const float* pos_glob    = (const float*)d_in[8];
    const float* w1_w        = (const float*)d_in[9];
    const float* w1_b        = (const float*)d_in[10];
    const float* w_2         = (const float*)d_in[11];
    const float* glu1_w      = (const float*)d_in[12];
    const float* glu1_b      = (const float*)d_in[13];
    const float* glu2_w      = (const float*)d_in[14];
    const int*   G_rows      = (const int*)d_in[15];
    const int*   G_cols      = (const int*)d_in[16];
    const float* G_vals      = (const float*)d_in[17];
    const int*   HG_rows     = (const int*)d_in[18];
    const int*   HG_cols     = (const int*)d_in[19];
    const float* HG_vals     = (const float*)d_in[20];
    const int*   seqs        = (const int*)d_in[21];
    const int*   masks       = (const int*)d_in[22];
    const float* geo_adjs    = (const float*)d_in[23];
    const int*   uidx        = (const int*)d_in[24];
    const float* lens        = (const float*)d_in[25];
    const int*   rev_seqs    = (const int*)d_in[26];

    float* out = (float*)d_out;

    const long NE  = (long)NNODES * EMBD;          // 7,680,128
    const long CE  = (long)(NNODES - 1) * EMBD;    // 7,680,000
    const long BLD = (long)NB * SEQ * EMBD;        // 13,107,200
    const long PF  = (long)NPOIS * EMBD;           // 6,400,000
    const long BL  = (long)NB * SEQ;               // 102,400
    const long PAD = (long)NB * 128 * 128;         // 16,777,216

    float* ws      = (float*)d_ws;
    float* nodes   = ws;
    float* acc     = nodes + NE;
    float* core    = acc + NE;          // phase 2: hs / hsg / beta
    float* bufA    = core + CE;         // seq_e -> mha_out -> seq_h
    float* bufB    = bufA + BLD;        // geo_raw -> tot -> nh1
    float* bufC    = bufB + BLD;        // geo_e -> attn_o -> fusion -> nh2
    float* bufQ    = bufC + BLD;        // seq_padT -> qkv -> g0/g1/g2 -> cat
    float* adj_pad = bufQ + 3 * BLD;    // persistent padded adjacency (f32)

    const int TPB = 256;
    auto g1d = [](long n) { return dim3((unsigned)((n + 255) / 256)); };

    // ---- one-time init ------------------------------------------------------
    init2_kernel<<<g1d(NE), TPB, 0, stream>>>(nodes, acc, nodes_emb, NE);
    adjpad_kernel<<<g1d(PAD), TPB, 0, stream>>>(geo_adjs, adj_pad);

    // ---- N_LOCAL = 2 spatio-temporal layers ---------------------------------
    for (int it = 0; it < 2; ++it) {
        gather_rows_kernel<<<g1d(BLD), TPB, 0, stream>>>(nodes, seqs, bufA, BL);

        // seq_padT[b] = seq_e[b]^T zero-padded to 128x128 (lives in bufQ, dead before qkv)
        seqT_kernel<<<g1d(PAD), TPB, 0, stream>>>(bufA, bufQ);

        // geo_raw[b](100x128) = adj_pad[b](128x128) @ seq_padT[b]^T   (store guard on M)
        wmma_gemm_kernel<0, 0, 1><<<dim3(4, 1, NB), TPB, 0, stream>>>(
            adj_pad, bufQ, nullptr, bufB, SEQ, EMBD, 128, 128, 128, EMBD,
            16384L, 16384L, (long)SEQ * EMBD, nullptr);

        // geo_e = relu(geo_raw @ fc_geo_w^T + b)
        wmma_gemm_kernel<1, 1, 0><<<dim3(4, 800, 1), TPB, 0, stream>>>(
            bufB, fc_geo_w, fc_geo_b, bufC, (int)BL, EMBD, EMBD, EMBD, EMBD, EMBD,
            0, 0, 0, nullptr);

        tot_kernel<<<g1d(BLD), TPB, 0, stream>>>(bufA, bufC, pos_local, bufB);

        // qkv = tot @ in_proj_w^T + b   (N = 384)
        wmma_gemm_kernel<0, 1, 0><<<dim3(12, 800, 1), TPB, 0, stream>>>(
            bufB, in_proj_w, in_proj_b, bufQ, (int)BL, 3 * EMBD, EMBD, EMBD, EMBD, 3 * EMBD,
            0, 0, 0, nullptr);

        attention_kernel<<<NB * NHEADS, 128, 0, stream>>>(bufQ, bufC);

        // mha = o @ out_proj_w^T + b
        wmma_gemm_kernel<0, 1, 0><<<dim3(4, 800, 1), TPB, 0, stream>>>(
            bufC, out_proj_w, out_proj_b, bufA, (int)BL, EMBD, EMBD, EMBD, EMBD, EMBD,
            0, 0, 0, nullptr);

        users_kernel<<<NB, EMBD, 0, stream>>>(bufA, uidx, nodes);

        zero_kernel<<<g1d(CE), TPB, 0, stream>>>(core, CE);
        spmm_kernel<<<g1d(NNZG * EMBD), TPB, 0, stream>>>(G_rows, G_cols, G_vals,
                                                          nodes, core, NNZG);
        commit_kernel<<<g1d(NE), TPB, 0, stream>>>(nodes, acc, core, NE);
    }

    // ---- global hypergraph branch (N_GLOBAL = 2) ----------------------------
    float* g0 = bufQ; float* g1 = bufQ + PF; float* g2 = bufQ + 2 * PF;
    copy_kernel<<<g1d(PF), TPB, 0, stream>>>(g0, nodes_emb + (long)NUSERS * EMBD, PF);
    copy_kernel<<<g1d(PF), TPB, 0, stream>>>(g2, g0, PF);

    zero_kernel<<<g1d(PF), TPB, 0, stream>>>(g1, PF);
    spmm_kernel<<<g1d(NNZH * EMBD), TPB, 0, stream>>>(HG_rows, HG_cols, HG_vals, g0, g1, NNZH);
    add_kernel<<<g1d(PF), TPB, 0, stream>>>(g2, g1, PF);

    zero_kernel<<<g1d(PF), TPB, 0, stream>>>(g0, PF);
    spmm_kernel<<<g1d(NNZH * EMBD), TPB, 0, stream>>>(HG_rows, HG_cols, HG_vals, g1, g0, NNZH);
    add_kernel<<<g1d(PF), TPB, 0, stream>>>(g2, g0, PF);

    // ---- fusion + pois output -----------------------------------------------
    fusion_kernel<<<g1d(NE), TPB, 0, stream>>>(acc, g2, bufC, out + (long)NB * EMBD, NE);

    // ---- gated selection head -----------------------------------------------
    gather_rows_kernel<<<g1d(BLD), TPB, 0, stream>>>(bufC, rev_seqs, bufA, BL);

    float* hs   = core;
    float* hsg  = core + (long)NB * EMBD;
    float* beta = core + 2L * NB * EMBD;
    hs_kernel<<<NB, EMBD, 0, stream>>>(bufA, lens, hs);

    cat_kernel<<<g1d(BLD), TPB, 0, stream>>>(pos_glob, masks, bufA, bufQ);

    // nh1 = tanh(cat(256) @ w1_w^T + b)
    wmma_gemm_kernel<2, 1, 0><<<dim3(4, 800, 1), TPB, 0, stream>>>(
        bufQ, w1_w, w1_b, bufB, (int)BL, EMBD, 2 * EMBD, 2 * EMBD, 2 * EMBD, EMBD,
        0, 0, 0, nullptr);

    // hsg = hs @ glu2_w^T
    wmma_gemm_kernel<0, 0, 0><<<dim3(4, 8, 1), TPB, 0, stream>>>(
        hs, glu2_w, nullptr, hsg, NB, EMBD, EMBD, EMBD, EMBD, EMBD, 0, 0, 0, nullptr);

    // nh2 = sigmoid(nh1 @ glu1_w^T + glu1_b + hsg[b])
    wmma_gemm_kernel<3, 1, 0><<<dim3(4, 800, 1), TPB, 0, stream>>>(
        bufB, glu1_w, glu1_b, bufC, (int)BL, EMBD, EMBD, EMBD, EMBD, EMBD,
        0, 0, 0, hsg);

    beta_kernel<<<g1d(BL), TPB, 0, stream>>>(bufC, w_2, beta, BL);
    select_kernel<<<NB, EMBD, 0, stream>>>(beta, bufA, acc, uidx, out);
}
